// CRF_979252544377
// MI455X (gfx1250) — matrix operand order repaired and checked by reading, not compiled, not run on hardware
//
#include <hip/hip_runtime.h>

typedef __attribute__((ext_vector_type(16))) _Float16 v16h;
typedef __attribute__((ext_vector_type(8)))  _Float16 v8h;
typedef __attribute__((ext_vector_type(8)))  float    v8f;

#define NSTATES 48
#define NFEAT   1024
#define BB      256
#define TT      512
#define NROWS   (BB*TT)        /* 131072 rows of the (B*T, F) matrix */
#define CH      32             /* recursion timesteps per LDS chunk  */
#define NCH     (TT/CH)        /* 16 chunks                          */

static __device__ __forceinline__ float clampf(float x, float lo, float hi) {
    return fminf(fmaxf(x, lo), hi);
}

#define WAIT_ASYNC() asm volatile("s_wait_asynccnt 0" ::: "memory")

/* ------------------------------------------------------------------ */
/* Kernel 1: per-(b,f) mean and inverse std over the T axis (ddof=1). */
/* ------------------------------------------------------------------ */
__global__ __launch_bounds__(256) void crf_stats(
    const float* __restrict__ feat,
    float* __restrict__ meanv,
    float* __restrict__ rstdv)
{
    const int col = blockIdx.x * blockDim.x + threadIdx.x;   /* b*NFEAT + f */
    const int b   = col >> 10;
    const int f   = col & (NFEAT - 1);
    const float* p = feat + ((size_t)b * TT) * NFEAT + f;

    float s = 0.f, s2 = 0.f;
    #pragma unroll 8
    for (int t = 0; t < TT; ++t) {
        const float x = p[(size_t)t * NFEAT];
        s  += x;
        s2  = fmaf(x, x, s2);
    }
    const float m   = s * (1.f / TT);
    const float var = (s2 - s * m) * (1.f / (TT - 1));
    const float sd  = sqrtf(fmaxf(var, 0.f));
    meanv[col] = m;
    rstdv[col] = (sd > 0.f) ? (1.f / sd) : 1.f;   /* std==0 -> treat as 1 */
}

/* ------------------------------------------------------------------ */
/* Kernel 2: sf = clip( clip((x-mu)*rstd,±10) @ W^T + bias, ±5 )      */
/* WMMA f32 <- f16 16x16x32. Weights converted to f16 ONCE per block  */
/* into LDS, pre-swizzled into per-(kc,nt,lane) 32B fragment slots so */
/* the K-loop does only ds_load_b128 for B.                           */
/* Dynamic LDS: 32*3*32 slots * 16 f16 = 49152 halfs = 96 KB.         */
/* ------------------------------------------------------------------ */
__global__ __launch_bounds__(128) void crf_gemm_sf(
    const float* __restrict__ feat,
    const float* __restrict__ meanv,
    const float* __restrict__ rstdv,
    const float* __restrict__ W,       /* (48, 1024) row-major */
    const float* __restrict__ sbias,   /* (48,) */
    float* __restrict__ sf)            /* (131072, 48) */
{
    extern __shared__ _Float16 ldsW[];             /* 49152 halfs */

    /* ---- one-time W -> f16 LDS staging, fragment-swizzled.
       slot s = (kc*3 + nt)*32 + lane holds the 16 halfs lane needs:
       n = nt*16 + (lane&15), k = kc*32 + (lane>>4)*8 + (i>>3)*16 + (i&7). */
    for (int s = threadIdx.x; s < 3072; s += blockDim.x) {
        const int kc = s / 96;
        const int r  = s % 96;
        const int nt = r >> 5;
        const int ln = r & 31;
        const int n  = nt * 16 + (ln & 15);
        const int kb = kc * 32 + (ln >> 4) * 8;
        const float* wp = W + (size_t)n * NFEAT + kb;
        #pragma unroll
        for (int q = 0; q < 2; ++q) {
            const float4 w0 = *(const float4*)(wp + q * 16);
            const float4 w1 = *(const float4*)(wp + q * 16 + 4);
            v8h h;
            h[0] = (_Float16)w0.x; h[1] = (_Float16)w0.y;
            h[2] = (_Float16)w0.z; h[3] = (_Float16)w0.w;
            h[4] = (_Float16)w1.x; h[5] = (_Float16)w1.y;
            h[6] = (_Float16)w1.z; h[7] = (_Float16)w1.w;
            *(v8h*)&ldsW[(size_t)s * 16 + q * 8] = h;
        }
    }
    __syncthreads();

    const int lane = threadIdx.x & 31;
    const int l15  = lane & 15;
    const int half = lane >> 4;
    const int wid  = blockIdx.x * (blockDim.x >> 5) + (threadIdx.x >> 5);
    const int m0   = wid << 4;               /* first row of this wave's tile */
    const int row  = m0 + l15;               /* row this lane loads A from    */
    const int bidx = row >> 9;               /* batch index = row / TT        */

    const float* frow = feat  + (size_t)row  * NFEAT;
    const float* mrow = meanv + (size_t)bidx * NFEAT;
    const float* rrow = rstdv + (size_t)bidx * NFEAT;

    v8f acc[3] = {};                         /* 3 N-tiles of f32 accum */

    for (int kc = 0; kc < NFEAT / 32; ++kc) {
        /* ---- A fragment: 16 f16 per lane, normalized on the fly.
           lane<16: M=l15, K = kc*32 + {0..7, 16..23}
           lane>=16: M=l15, K = kc*32 + {8..15, 24..31}                */
        v16h afrag;
        #pragma unroll
        for (int g = 0; g < 2; ++g) {
            const int kb = kc * 32 + half * 8 + g * 16;
            const float4 x0  = *(const float4*)(frow + kb);
            const float4 x1  = *(const float4*)(frow + kb + 4);
            const float4 mu0 = *(const float4*)(mrow + kb);
            const float4 mu1 = *(const float4*)(mrow + kb + 4);
            const float4 rs0 = *(const float4*)(rrow + kb);
            const float4 rs1 = *(const float4*)(rrow + kb + 4);
            afrag[g*8 + 0] = (_Float16)clampf((x0.x - mu0.x) * rs0.x, -10.f, 10.f);
            afrag[g*8 + 1] = (_Float16)clampf((x0.y - mu0.y) * rs0.y, -10.f, 10.f);
            afrag[g*8 + 2] = (_Float16)clampf((x0.z - mu0.z) * rs0.z, -10.f, 10.f);
            afrag[g*8 + 3] = (_Float16)clampf((x0.w - mu0.w) * rs0.w, -10.f, 10.f);
            afrag[g*8 + 4] = (_Float16)clampf((x1.x - mu1.x) * rs1.x, -10.f, 10.f);
            afrag[g*8 + 5] = (_Float16)clampf((x1.y - mu1.y) * rs1.y, -10.f, 10.f);
            afrag[g*8 + 6] = (_Float16)clampf((x1.z - mu1.z) * rs1.z, -10.f, 10.f);
            afrag[g*8 + 7] = (_Float16)clampf((x1.w - mu1.w) * rs1.w, -10.f, 10.f);
        }

        /* ---- B fragments straight out of LDS (2x ds_load_b128 each) */
        #pragma unroll
        for (int nt = 0; nt < 3; ++nt) {
            const int slot = (kc * 3 + nt) * 32 + lane;
            const v8h b0 = *(const v8h*)&ldsW[(size_t)slot * 16];
            const v8h b1 = *(const v8h*)&ldsW[(size_t)slot * 16 + 8];
            v16h bfrag;
            #pragma unroll
            for (int i = 0; i < 8; ++i) { bfrag[i] = b0[i]; bfrag[i + 8] = b1[i]; }
            acc[nt] = __builtin_amdgcn_wmma_f32_16x16x32_f16(
                false, afrag, false, bfrag, (short)0, acc[nt], false, false);
        }
    }

    /* ---- Epilogue: C layout 16x16 f32 -> VGPR r: M = r + 8*half, N = l15 */
    #pragma unroll
    for (int nt = 0; nt < 3; ++nt) {
        const float bv = sbias[nt * 16 + l15];
        #pragma unroll
        for (int r = 0; r < 8; ++r) {
            const int orow = m0 + r + half * 8;
            sf[(size_t)orow * NSTATES + nt * 16 + l15] =
                clampf(acc[nt][r] + bv, -5.f, 5.f);
        }
    }
}

/* ------------------------------------------------------------------ */
/* Kernel 3: forward recursion, one block (64 threads) per batch.     */
/*   alpha_t[k] = log( sum_j exp(aprev[j]-M) * E[j,k] ) + M + sf[t,k] */
/* where E = exp(clip(w+b,±5)) is precomputed into 48 registers per   */
/* thread (column k). Per step: 1 exp + 48 FMA + 1 log.               */
/* sf is streamed through LDS in 32-step chunks, double buffered via  */
/* global_load_async_to_lds_b128 (ASYNCcnt).                          */
/* ------------------------------------------------------------------ */
__global__ __launch_bounds__(64) void crf_forward(
    const float* __restrict__ sf,
    const float* __restrict__ tw,
    const float* __restrict__ tb,
    float* __restrict__ out)           /* [0,256): logZ ; [256,...): alpha */
{
    __shared__ float sfchunk[2][CH * NSTATES];     /* 12 KB double buffer */
    __shared__ float alpha[2][NSTATES];
    __shared__ float evec[NSTATES];

    const int b   = blockIdx.x;
    const int k   = threadIdx.x;
    const int tid = threadIdx.x;

    const float* sfb  = sf + (size_t)b * TT * NSTATES;
    float*       aout = out + BB + (size_t)b * TT * NSTATES;

    /* exp(trans) column k -> registers (only threads k<48 use it) */
    float etw[NSTATES];
    if (k < NSTATES) {
        #pragma unroll 8
        for (int j = 0; j < NSTATES; ++j)
            etw[j] = __expf(clampf(tw[j * NSTATES + k] + tb[j * NSTATES + k],
                                   -5.f, 5.f));
    }

    /* async chunk loader: 1536 floats = 384 b128 slots, 6 per thread */
    auto issue_chunk = [&](int c) {
        const float* src = sfb + (size_t)c * CH * NSTATES;
        float* dst = sfchunk[c & 1];
        #pragma unroll
        for (int i = 0; i < 6; ++i) {
            const int slot = tid + i * 64;
            unsigned           loff = (unsigned)(size_t)(const void*)(dst + slot * 4);
            unsigned long long ga   = (unsigned long long)(const void*)(src + slot * 4);
            asm volatile("global_load_async_to_lds_b128 %0, %1, off"
                         :: "v"(loff), "v"(ga) : "memory");
        }
    };

    issue_chunk(0);
    WAIT_ASYNC();
    __syncthreads();

    /* t = 0 */
    if (k < NSTATES) {
        const float a0 = sfchunk[0][k];
        alpha[0][k] = a0;
        aout[k]     = a0;
    }
    __syncthreads();

    float M = -3.4e38f;
    #pragma unroll
    for (int j = 0; j < NSTATES; ++j) M = fmaxf(M, alpha[0][j]);

    int t = 1;
    for (int c = 0; c < NCH; ++c) {
        if (c + 1 < NCH) issue_chunk(c + 1);     /* prefetch next chunk */
        const float* sl = sfchunk[c & 1];
        const int i0 = (c == 0) ? 1 : 0;

        for (int i = i0; i < CH; ++i, ++t) {
            const float* ap = alpha[(t + 1) & 1];
            float*       an = alpha[t & 1];

            if (k < NSTATES) evec[k] = __expf(ap[k] - M);
            __syncthreads();

            if (k < NSTATES) {
                float s = 0.f;
                #pragma unroll
                for (int j = 0; j < NSTATES; ++j)
                    s = fmaf(evec[j], etw[j], s);
                const float a = __logf(s) + M + sl[i * NSTATES + k];
                an[k] = a;
                aout[(size_t)t * NSTATES + k] = a;
            }
            __syncthreads();

            float m = -3.4e38f;
            #pragma unroll
            for (int j = 0; j < NSTATES; ++j) m = fmaxf(m, an[j]);
            M = m;
        }

        if (c + 1 < NCH) WAIT_ASYNC();           /* next chunk landed */
        __syncthreads();
    }

    /* log partition over alpha[T-1] (parity (TT-1)&1 == 1) */
    if (k == 0) {
        const float* af = alpha[(TT - 1) & 1];
        float s = 0.f;
        for (int j = 0; j < NSTATES; ++j) s += __expf(af[j] - M);
        out[b] = __logf(s) + M;
    }
}

/* ------------------------------------------------------------------ */
extern "C" void kernel_launch(void* const* d_in, const int* in_sizes, int n_in,
                              void* d_out, int out_size, void* d_ws, size_t ws_size,
                              hipStream_t stream)
{
    const float* feat  = (const float*)d_in[0];   /* (256,512,1024) */
    const float* W     = (const float*)d_in[1];   /* (48,1024)      */
    const float* sbias = (const float*)d_in[2];   /* (48,)          */
    const float* tw    = (const float*)d_in[3];   /* (48,48)        */
    const float* tb    = (const float*)d_in[4];   /* (48,48)        */
    float* out = (float*)d_out;

    /* workspace layout: mean (1 MB) | rstd (1 MB) | sf (25.2 MB) */
    float* meanv = (float*)d_ws;
    float* rstdv = meanv + (size_t)BB * NFEAT;
    float* sfbuf = rstdv + (size_t)BB * NFEAT;

    crf_stats<<<(BB * NFEAT) / 256, 256, 0, stream>>>(feat, meanv, rstdv);

    /* 131072 rows / (16 rows/wave * 4 waves/block) = 2048 blocks,
       96 KB dynamic LDS for the f16 weight fragments */
    crf_gemm_sf<<<NROWS / 64, 128, 49152 * sizeof(_Float16), stream>>>(
        feat, meanv, rstdv, W, sbias, sfbuf);

    crf_forward<<<BB, 64, 0, stream>>>(sfbuf, tw, tb, out);
}
